// TransformerEncoderLayer_63436666962102
// MI455X (gfx1250) — compile-verified, matrix operand order checked by reading, not run on hardware
//
#include <hip/hip_runtime.h>

typedef __bf16 bf16_t;
typedef bf16_t v8bf  __attribute__((ext_vector_type(8)));
typedef bf16_t v16bf __attribute__((ext_vector_type(16)));
typedef float  v8f   __attribute__((ext_vector_type(8)));

#define B_   128
#define S_   128
#define D_   1024
#define F_   4096
#define ROWS (B_ * S_)
#define NHEAD 16
#define DH    64

static __device__ __forceinline__ v16bf cat8(v8bf lo, v8bf hi) {
  return __builtin_shufflevector(lo, hi, 0,1,2,3,4,5,6,7,8,9,10,11,12,13,14,15);
}

// CDNA5 async global->LDS copy (16B per lane), tracked by ASYNCcnt.
static __device__ __forceinline__ void async_ld16(void* ldsPtr, const void* gPtr) {
  unsigned int ldsOff = (unsigned int)(unsigned long long)ldsPtr; // addr[31:0] = LDS byte offset
  asm volatile("global_load_async_to_lds_b128 %0, %1, off"
               :: "v"(ldsOff), "v"((unsigned long long)gPtr)
               : "memory");
}
static __device__ __forceinline__ void wait_async0() {
  asm volatile("s_wait_asynccnt 0" ::: "memory");
}

// ---------------------------------------------------------------------------
// Weight convert + transpose: W f32 [K][N] -> Wt bf16 [N][K]
// ---------------------------------------------------------------------------
__global__ __launch_bounds__(256)
void wconv_kernel(bf16_t* __restrict__ dst, const float* __restrict__ src,
                  int K, int N) {
  long i = (long)blockIdx.x * 256 + threadIdx.x;
  long k = i / N, n = i % N;
  dst[n * (long)K + k] = (bf16_t)src[i];
}

// ---------------------------------------------------------------------------
// Strided per-batch f32 copy (float4), optional bf16 mirror.
// ---------------------------------------------------------------------------
__global__ __launch_bounds__(256)
void copy_seg_kernel(float* __restrict__ dst, bf16_t* __restrict__ dstb,
                     const float* __restrict__ src,
                     long dstStride, long dstOff, long srcStride, long srcOff,
                     int perBatch4) {
  long i = (long)blockIdx.x * 256 + threadIdx.x;
  long b = i / perBatch4, r = (i % perBatch4) * 4;
  float4 v = *(const float4*)(src + b * srcStride + srcOff + r);
  *(float4*)(dst + b * dstStride + dstOff + r) = v;
  if (dstb) {
    bf16_t* d = dstb + b * dstStride + dstOff + r;
    d[0] = (bf16_t)v.x; d[1] = (bf16_t)v.y; d[2] = (bf16_t)v.z; d[3] = (bf16_t)v.w;
  }
}

// ---------------------------------------------------------------------------
// WMMA bf16 GEMM:  C[rows][N] = act(A[rows][K] @ Wt^T + bias) (+ residual)
//   A bf16 [rows][K], Wt bf16 [N][K].  Output f32 or bf16.
//   Block tile 128x128, BK=32, double-buffered async LDS staging.
//   8 waves (4x2), wave tile 32x64 -> 8 WMMA per k-tile per wave.
// ---------------------------------------------------------------------------
template<int OBF, int RELU, int RES>
__global__ __launch_bounds__(256)
void gemm_wmma(const bf16_t* __restrict__ A, const bf16_t* __restrict__ Bt,
               const float* __restrict__ bias, void* __restrict__ Cp,
               const float* __restrict__ res, int K, int N) {
  constexpr int BM = 128, BN = 128, BK = 32, LDT = BK + 8;
  __shared__ bf16_t sA[2][BM * LDT];
  __shared__ bf16_t sB[2][BN * LDT];

  const int tid  = threadIdx.x;
  const int lane = tid & 31;
  const int wave = tid >> 5;
  const int wm = wave >> 1, wn = wave & 1;
  const int r15 = lane & 15;
  const int kb  = (lane & 16) ? 8 : 0;
  const long m0 = (long)blockIdx.x * BM;
  const long n0 = (long)blockIdx.y * BN;

  auto stage = [&](int buf, int kt) {
    #pragma unroll
    for (int it = 0; it < 2; ++it) {
      int i = tid + it * 256;               // 512 x 16B per operand tile
      int row = i >> 2, col = (i & 3) * 8;
      async_ld16(&sA[buf][row * LDT + col], A  + (m0 + row) * (long)K + kt + col);
      async_ld16(&sB[buf][row * LDT + col], Bt + (n0 + row) * (long)K + kt + col);
    }
  };

  v8f acc[2][4] = {};
  const int nk = K / BK;
  stage(0, 0);
  int cur = 0;
  for (int t = 0; t < nk; ++t) {
    wait_async0();
    __syncthreads();                        // tile t resident in buf 'cur'
    if (t + 1 < nk) stage(cur ^ 1, (t + 1) * BK);

    v16bf a[2], b[4];
    #pragma unroll
    for (int mi = 0; mi < 2; ++mi) {
      const bf16_t* p = &sA[cur][(wm * 32 + mi * 16 + r15) * LDT];
      a[mi] = cat8(*(const v8bf*)(p + kb), *(const v8bf*)(p + 16 + kb));
    }
    #pragma unroll
    for (int ni = 0; ni < 4; ++ni) {
      const bf16_t* p = &sB[cur][(wn * 64 + ni * 16 + r15) * LDT];
      b[ni] = cat8(*(const v8bf*)(p + kb), *(const v8bf*)(p + 16 + kb));
    }
    #pragma unroll
    for (int mi = 0; mi < 2; ++mi)
      #pragma unroll
      for (int ni = 0; ni < 4; ++ni)
        acc[mi][ni] = __builtin_amdgcn_wmma_f32_16x16x32_bf16(
            false, a[mi], false, b[ni], (short)0, acc[mi][ni], false, false);
    cur ^= 1;
  }

  const int half8 = (lane & 16) ? 8 : 0;
  #pragma unroll
  for (int mi = 0; mi < 2; ++mi)
    #pragma unroll
    for (int ni = 0; ni < 4; ++ni) {
      long gcol = n0 + wn * 64 + ni * 16 + r15;
      float bv = bias[gcol];
      #pragma unroll
      for (int i = 0; i < 8; ++i) {
        long grow = m0 + wm * 32 + mi * 16 + i + half8;
        float v = acc[mi][ni][i] + bv;
        if constexpr (RES) v += res[grow * (long)N + gcol];
        if constexpr (RELU) v = fmaxf(v, 0.0f);
        if constexpr (OBF)
          ((bf16_t*)Cp)[grow * (long)N + gcol] = (bf16_t)v;
        else
          ((float*)Cp)[grow * (long)N + gcol] = v;
      }
    }
}

// ---------------------------------------------------------------------------
// Attention: one block per (batch, head).  Q,K,V,O bf16 [B*S][D], head slice
// at column h*64.  S = softmax(QK^T/8) in registers, O = S@V, all WMMA bf16.
// ---------------------------------------------------------------------------
__global__ __launch_bounds__(256)
void attn_kernel(const bf16_t* __restrict__ gQ, const bf16_t* __restrict__ gK,
                 const bf16_t* __restrict__ gV, bf16_t* __restrict__ gO) {
  constexpr int LDV = S_ + 8;
  constexpr int LDP = S_ + 8;
  __shared__ bf16_t sVt[DH * LDV];
  __shared__ bf16_t sP[S_ * LDP];

  const int tid  = threadIdx.x;
  const int lane = tid & 31;
  const int wave = tid >> 5;
  const int r15 = lane & 15;
  const int kb  = (lane & 16) ? 8 : 0;
  const int half8 = (lane & 16) ? 8 : 0;
  const int b = blockIdx.x >> 4;
  const int h = blockIdx.x & 15;
  const long base = (long)b * S_ * D_ + h * DH;

  // V[seq][dh] -> Vt[dh][seq]
  for (int i = tid; i < S_ * DH; i += 256) {
    int s = i >> 6, dh = i & 63;
    sVt[dh * LDV + s] = gV[base + (long)s * D_ + dh];
  }

  // ---- S = Q @ K^T ----
  const int qrow = wave * 16 + r15;
  v16bf aq[2];
  #pragma unroll
  for (int ks = 0; ks < 2; ++ks) {
    const bf16_t* p = gQ + base + (long)qrow * D_ + ks * 32;
    aq[ks] = cat8(*(const v8bf*)(p + kb), *(const v8bf*)(p + 16 + kb));
  }
  v8f acc[8] = {};
  #pragma unroll
  for (int t = 0; t < 8; ++t) {
    const int krow = t * 16 + r15;
    #pragma unroll
    for (int ks = 0; ks < 2; ++ks) {
      const bf16_t* p = gK + base + (long)krow * D_ + ks * 32;
      v16bf bk = cat8(*(const v8bf*)(p + kb), *(const v8bf*)(p + 16 + kb));
      acc[t] = __builtin_amdgcn_wmma_f32_16x16x32_bf16(
          false, aq[ks], false, bk, (short)0, acc[t], false, false);
    }
  }

  // ---- softmax (registers + 16-lane butterflies) ----
  const float scale = 0.125f;   // 1/sqrt(64)
  float rmax[8], rsum[8];
  #pragma unroll
  for (int i = 0; i < 8; ++i) {
    float m = -3.0e38f;
    #pragma unroll
    for (int t = 0; t < 8; ++t) {
      float v = acc[t][i] * scale;
      acc[t][i] = v;
      m = fmaxf(m, v);
    }
    #pragma unroll
    for (int off = 1; off < 16; off <<= 1)
      m = fmaxf(m, __shfl_xor(m, off, 32));
    rmax[i] = m;
  }
  #pragma unroll
  for (int i = 0; i < 8; ++i) {
    float s = 0.0f;
    #pragma unroll
    for (int t = 0; t < 8; ++t) {
      float e = __expf(acc[t][i] - rmax[i]);
      acc[t][i] = e;
      s += e;
    }
    #pragma unroll
    for (int off = 1; off < 16; off <<= 1)
      s += __shfl_xor(s, off, 32);
    rsum[i] = 1.0f / s;
  }

  // ---- P (bf16) to LDS ----
  #pragma unroll
  for (int t = 0; t < 8; ++t)
    #pragma unroll
    for (int i = 0; i < 8; ++i) {
      int row = wave * 16 + i + half8;
      int col = t * 16 + r15;
      sP[row * LDP + col] = (bf16_t)(acc[t][i] * rsum[i]);
    }
  __syncthreads();

  // ---- O = P @ V ----
  v8f oacc[4] = {};
  const int prow = wave * 16 + r15;
  #pragma unroll
  for (int ks = 0; ks < 4; ++ks) {
    const bf16_t* pp = &sP[prow * LDP + ks * 32];
    v16bf pa = cat8(*(const v8bf*)(pp + kb), *(const v8bf*)(pp + 16 + kb));
    #pragma unroll
    for (int nt = 0; nt < 4; ++nt) {
      const bf16_t* vp = &sVt[(nt * 16 + r15) * LDV + ks * 32];
      v16bf vb = cat8(*(const v8bf*)(vp + kb), *(const v8bf*)(vp + 16 + kb));
      oacc[nt] = __builtin_amdgcn_wmma_f32_16x16x32_bf16(
          false, pa, false, vb, (short)0, oacc[nt], false, false);
    }
  }
  #pragma unroll
  for (int nt = 0; nt < 4; ++nt)
    #pragma unroll
    for (int i = 0; i < 8; ++i) {
      int row = wave * 16 + i + half8;
      int col = nt * 16 + r15;
      gO[base + (long)row * D_ + col] = (bf16_t)oacc[nt][i];
    }
}

// ---------------------------------------------------------------------------
// LayerNorm over entire (S_ x D_) per batch, f32 + bf16 outputs.
// ---------------------------------------------------------------------------
__global__ __launch_bounds__(256)
void ln_kernel(const float* __restrict__ x, const float* __restrict__ g,
               const float* __restrict__ beta, float* __restrict__ dst,
               bf16_t* __restrict__ dstb) {
  constexpr int NEL = S_ * D_;
  __shared__ float red[256], red2[256];
  const int tid = threadIdx.x;
  const long base = (long)blockIdx.x * NEL;

  float s = 0.0f, s2 = 0.0f;
  for (int i = tid; i < NEL; i += 256) {
    float v = x[base + i];
    s += v; s2 += v * v;
  }
  red[tid] = s; red2[tid] = s2;
  __syncthreads();
  for (int o = 128; o > 0; o >>= 1) {
    if (tid < o) { red[tid] += red[tid + o]; red2[tid] += red2[tid + o]; }
    __syncthreads();
  }
  const float inv = 1.0f / (float)NEL;
  const float mu = red[0] * inv;
  const float var = red2[0] * inv - mu * mu;
  const float rs = rsqrtf(var + 1e-5f);
  for (int i = tid; i < NEL; i += 256) {
    float v = (x[base + i] - mu) * rs * g[i] + beta[i];
    dst[base + i] = v;
    if (dstb) dstb[base + i] = (bf16_t)v;
  }
}

// ---------------------------------------------------------------------------
// Host orchestration
// ---------------------------------------------------------------------------
extern "C" void kernel_launch(void* const* d_in, const int* in_sizes, int n_in,
                              void* d_out, int out_size, void* d_ws, size_t ws_size,
                              hipStream_t stream) {
  const float* src = (const float*)d_in[0];
  const float* Wq = (const float*)d_in[1];  const float* bq = (const float*)d_in[2];
  const float* Wk = (const float*)d_in[3];  const float* bk = (const float*)d_in[4];
  const float* Wv = (const float*)d_in[5];  const float* bv = (const float*)d_in[6];
  const float* Wo = (const float*)d_in[7];  const float* bo = (const float*)d_in[8];
  const float* W1 = (const float*)d_in[9];  const float* b1 = (const float*)d_in[10];
  const float* W2 = (const float*)d_in[11]; const float* b2 = (const float*)d_in[12];
  const float* g1 = (const float*)d_in[13]; const float* beta1 = (const float*)d_in[14];
  const float* g2 = (const float*)d_in[15]; const float* beta2 = (const float*)d_in[16];
  float* out = (float*)d_out;

  char* p = (char*)d_ws;
  auto carve = [&](size_t bytes) -> char* {
    char* r = p; p += (bytes + 255) & ~(size_t)255; return r;
  };
  bf16_t* wtq = (bf16_t*)carve((size_t)D_ * D_ * 2);
  bf16_t* wtk = (bf16_t*)carve((size_t)D_ * D_ * 2);
  bf16_t* wtv = (bf16_t*)carve((size_t)D_ * D_ * 2);
  bf16_t* wto = (bf16_t*)carve((size_t)D_ * D_ * 2);
  bf16_t* wt1 = (bf16_t*)carve((size_t)D_ * F_ * 2);
  bf16_t* wt2 = (bf16_t*)carve((size_t)F_ * D_ * 2);
  float*  x1  = (float*)carve((size_t)ROWS * D_ * 4);
  float*  x2  = (float*)carve((size_t)ROWS * D_ * 4);
  bf16_t* x1b = (bf16_t*)carve((size_t)ROWS * D_ * 2);
  bf16_t* x2b = (bf16_t*)carve((size_t)ROWS * D_ * 2);
  bf16_t* Qb  = (bf16_t*)carve((size_t)ROWS * D_ * 2);
  bf16_t* Kb  = (bf16_t*)carve((size_t)ROWS * D_ * 2);
  bf16_t* Vb  = (bf16_t*)carve((size_t)ROWS * D_ * 2);
  bf16_t* AOb = (bf16_t*)carve((size_t)ROWS * D_ * 2);
  float*  yf  = (float*)carve((size_t)ROWS * D_ * 4);
  float*  t1  = (float*)carve((size_t)ROWS * D_ * 4);
  bf16_t* t1b = (bf16_t*)carve((size_t)ROWS * D_ * 2);
  bf16_t* Hb  = (bf16_t*)carve((size_t)ROWS * F_ * 2);
  float*  m1  = (float*)carve((size_t)ROWS * D_ * 4);
  bf16_t* m1b = (bf16_t*)carve((size_t)ROWS * D_ * 2);
  float*  m2  = (float*)carve((size_t)ROWS * D_ * 4);
  bf16_t* m2b = (bf16_t*)carve((size_t)ROWS * D_ * 2);
  (void)ws_size; (void)in_sizes; (void)n_in; (void)out_size;

  // ---- weight convert + transpose ----
  wconv_kernel<<<(D_ * D_) / 256, 256, 0, stream>>>(wtq, Wq, D_, D_);
  wconv_kernel<<<(D_ * D_) / 256, 256, 0, stream>>>(wtk, Wk, D_, D_);
  wconv_kernel<<<(D_ * D_) / 256, 256, 0, stream>>>(wtv, Wv, D_, D_);
  wconv_kernel<<<(D_ * D_) / 256, 256, 0, stream>>>(wto, Wo, D_, D_);
  wconv_kernel<<<(D_ * F_) / 256, 256, 0, stream>>>(wt1, W1, D_, F_);
  wconv_kernel<<<(F_ * D_) / 256, 256, 0, stream>>>(wt2, W2, F_, D_);

  // ---- split src into modal1 / modal2 (f32 + bf16 mirror) ----
  const int pb4 = (S_ * D_) / 4;
  const int copyBlocks = (B_ * pb4) / 256;
  copy_seg_kernel<<<copyBlocks, 256, 0, stream>>>(x1, x1b, src, (long)S_ * D_, 0,
                                                  (long)2 * S_ * D_, 0, pb4);
  copy_seg_kernel<<<copyBlocks, 256, 0, stream>>>(x2, x2b, src, (long)S_ * D_, 0,
                                                  (long)2 * S_ * D_, (long)S_ * D_, pb4);

  const dim3 gD(ROWS / 128, D_ / 128);   // (128, 8)
  const dim3 gF(ROWS / 128, F_ / 128);   // (128, 32)

  auto run_pass = [&](const bf16_t* qxb, const float* qxf, const bf16_t* kvxb,
                      float* mOut, bf16_t* mOutB) {
    gemm_wmma<1, 0, 0><<<gD, 256, 0, stream>>>(qxb,  wtq, bq, Qb, nullptr, D_, D_);
    gemm_wmma<1, 0, 0><<<gD, 256, 0, stream>>>(kvxb, wtk, bk, Kb, nullptr, D_, D_);
    gemm_wmma<1, 0, 0><<<gD, 256, 0, stream>>>(kvxb, wtv, bv, Vb, nullptr, D_, D_);
    attn_kernel<<<B_ * NHEAD, 256, 0, stream>>>(Qb, Kb, Vb, AOb);
    gemm_wmma<0, 0, 1><<<gD, 256, 0, stream>>>(AOb, wto, bo, yf, qxf, D_, D_);
    ln_kernel<<<B_, 256, 0, stream>>>(yf, g1, beta1, t1, t1b);
    gemm_wmma<1, 1, 0><<<gF, 256, 0, stream>>>(t1b, wt1, b1, Hb, nullptr, D_, F_);
    gemm_wmma<0, 0, 1><<<gD, 256, 0, stream>>>(Hb, wt2, b2, yf, t1, F_, D_);
    ln_kernel<<<B_, 256, 0, stream>>>(yf, g2, beta2, mOut, mOutB);
  };

  run_pass(x1b, x1, x2b, m1, m1b);   // modal1 attends over modal2
  run_pass(x2b, x2, m1b, m2, m2b);   // modal2 attends over updated m1

  // ---- concat [m1, m2] into d_out ----
  copy_seg_kernel<<<copyBlocks, 256, 0, stream>>>(out, nullptr, m1, (long)2 * S_ * D_, 0,
                                                  (long)S_ * D_, 0, pb4);
  copy_seg_kernel<<<copyBlocks, 256, 0, stream>>>(out, nullptr, m2, (long)2 * S_ * D_,
                                                  (long)S_ * D_, (long)S_ * D_, 0, pb4);
}
